// fA_P4DenseBlock2D_78572131713150
// MI455X (gfx1250) — matrix operand order repaired and checked by reading, not compiled, not run on hardware
//
#include <hip/hip_runtime.h>
#include <hip/hip_bf16.h>

typedef __attribute__((ext_vector_type(16))) __bf16        v16bf;
typedef __attribute__((ext_vector_type(8)))  float         v8f;
typedef __attribute__((ext_vector_type(4)))  unsigned int  v4u;

#define R4    4
#define B_    16
#define C_    96
#define O_    48
#define H_    66
#define HW    (H_ * H_)        // 4356
#define OH    64
#define OPIX  (OH * OH)        // 4096
#define K_TOT (C_ * R4 * 9)    // 3456
#define EPSB  2e-5f

__device__ __forceinline__ unsigned short f2bf(float f) {
    unsigned int u = __float_as_uint(f);
    return (unsigned short)((u + 0x7FFFu + ((u >> 16) & 1u)) >> 16);  // RNE
}
__device__ __forceinline__ float sigmoidf_(float v) {
    return 1.0f / (1.0f + __expf(-v));
}
// Wave-relative LDS byte offset = low 32 bits of the generic pointer (LDS aperture).
__device__ __forceinline__ unsigned lds_off(const void* p) {
    return (unsigned)(unsigned long long)p;
}
// CDNA5 async copy: 16B global -> LDS, tracked by ASYNCcnt (no VGPR round-trip).
__device__ __forceinline__ void async_copy16(unsigned lds_byte, const void* g) {
    asm volatile("global_load_async_to_lds_b128 %0, %1, off"
                 :: "v"(lds_byte), "v"(g) : "memory");
}
__device__ __forceinline__ void wait_async0() {
#if __has_builtin(__builtin_amdgcn_s_wait_asynccnt)
    __builtin_amdgcn_s_wait_asynccnt(0);
#else
    asm volatile("s_wait_asynccnt 0x0" ::: "memory");
#endif
}

// ---------------- 1) BatchNorm statistics: one block per channel ----------------
__global__ void k_bn_stats(const float* __restrict__ x, float* __restrict__ mu, float* __restrict__ rs) {
    const int c = blockIdx.x;
    const int N = B_ * R4 * HW;
    float s = 0.f, s2 = 0.f;
    for (int i = threadIdx.x; i < N; i += 256) {
        int b   = i / (R4 * HW);
        int rem = i - b * (R4 * HW);
        float v = x[((size_t)(b * C_ + c) * R4) * HW + rem];
        s += v; s2 += v * v;
    }
    __shared__ float sh[256], sh2[256];
    sh[threadIdx.x] = s; sh2[threadIdx.x] = s2;
    __syncthreads();
    for (int off = 128; off > 0; off >>= 1) {
        if (threadIdx.x < off) {
            sh[threadIdx.x]  += sh[threadIdx.x + off];
            sh2[threadIdx.x] += sh2[threadIdx.x + off];
        }
        __syncthreads();
    }
    if (threadIdx.x == 0) {
        float m = sh[0] / (float)N;
        float var = sh2[0] / (float)N - m * m;
        mu[c] = m;
        rs[c] = rsqrtf(var + EPSB);
    }
}

// ---------------- 2) BN+ReLU (materialize y) + per-(b,c,r) spatial mean/max ----------------
__global__ void k_bn_relu_pool(const float* __restrict__ x, const float* __restrict__ mu,
                               const float* __restrict__ rs, const float* __restrict__ gamma,
                               const float* __restrict__ beta, float* __restrict__ y,
                               float* __restrict__ s_avg, float* __restrict__ s_max) {
    const int bid = blockIdx.x;            // b*C_*R4 + c*R4 + r
    const int c = (bid >> 2) % C_;
    const float* xp = x + (size_t)bid * HW;
    float* yp = y + (size_t)bid * HW;
    const float g = gamma[c] * rs[c];
    const float bt = beta[c] - mu[c] * g;
    float s = 0.f, mx = -1e30f;
    for (int i = threadIdx.x; i < HW; i += 256) {
        float v = fmaxf(xp[i] * g + bt, 0.f);
        yp[i] = v;
        s += v; mx = fmaxf(mx, v);
    }
    __shared__ float sh[256], shm[256];
    sh[threadIdx.x] = s; shm[threadIdx.x] = mx;
    __syncthreads();
    for (int off = 128; off > 0; off >>= 1) {
        if (threadIdx.x < off) {
            sh[threadIdx.x]  += sh[threadIdx.x + off];
            shm[threadIdx.x]  = fmaxf(shm[threadIdx.x], shm[threadIdx.x + off]);
        }
        __syncthreads();
    }
    if (threadIdx.x == 0) {
        s_avg[bid] = sh[0] / (float)HW;
        s_max[bid] = shm[0];
    }
}

// ---------------- 3) channel pooling: u_avg/u_max [B][R][HW] ----------------
__global__ void k_channel_pool(const float* __restrict__ y, float* __restrict__ u_avg,
                               float* __restrict__ u_max) {
    const int i = blockIdx.x * 256 + threadIdx.x;
    const int N = B_ * R4 * HW;
    if (i >= N) return;
    int b   = i / (R4 * HW);
    int rhw = i - b * (R4 * HW);
    const float* yp = y + (size_t)b * C_ * R4 * HW + rhw;
    float s = 0.f, mx = -1e30f;
    for (int c = 0; c < C_; ++c) {
        float v = yp[(size_t)c * R4 * HW];
        s += v; mx = fmaxf(mx, v);
    }
    u_avg[i] = s * (1.0f / (float)C_);
    u_max[i] = mx;
}

// ---------------- 4) channel attention MLP: one block per (b,h) ----------------
__global__ void k_channel_att(const float* __restrict__ s_avg, const float* __restrict__ s_max,
                              const float* __restrict__ fc1, const float* __restrict__ fc2,
                              float* __restrict__ ch_att) {
    const int b = blockIdx.x >> 2;
    const int h = blockIdx.x & 3;
    const float* sA = s_avg + b * C_ * R4;
    const float* sM = s_max + b * C_ * R4;
    float pa0 = 0.f, pa1 = 0.f, pm0 = 0.f, pm1 = 0.f;
    for (int cr = threadIdx.x; cr < C_ * R4; cr += 256) {
        int c = cr >> 2, r = cr & 3, rr = (r - h) & 3;
        float f0 = fc1[(0 * C_ + c) * R4 + rr];
        float f1 = fc1[(1 * C_ + c) * R4 + rr];
        float va = sA[cr], vm = sM[cr];
        pa0 += va * f0; pa1 += va * f1;
        pm0 += vm * f0; pm1 += vm * f1;
    }
    __shared__ float red[4][256];
    red[0][threadIdx.x] = pa0; red[1][threadIdx.x] = pa1;
    red[2][threadIdx.x] = pm0; red[3][threadIdx.x] = pm1;
    __syncthreads();
    for (int off = 128; off > 0; off >>= 1) {
        if (threadIdx.x < off)
            for (int q = 0; q < 4; ++q) red[q][threadIdx.x] += red[q][threadIdx.x + off];
        __syncthreads();
    }
    const float h0 = fmaxf(red[0][0], 0.f) + fmaxf(red[2][0], 0.f);
    const float h1 = fmaxf(red[1][0], 0.f) + fmaxf(red[3][0], 0.f);
    for (int cr = threadIdx.x; cr < C_ * R4; cr += 256) {
        int c = cr >> 2, r = cr & 3, rr = (r - h) & 3;
        float v = h0 * fc2[(c * 2 + 0) * R4 + rr] + h1 * fc2[(c * 2 + 1) * R4 + rr];
        ch_att[(size_t)(b * R4 + h) * (C_ * R4) + cr] = sigmoidf_(v);
    }
}

// ---------------- 5) rotate/roll the 7x7 spatial-attention kernel: sa_h[4][8][49] ----------------
__global__ void k_sa_transform(const float* __restrict__ sa_w, float* __restrict__ sa_h) {
    const int i = blockIdx.x * 256 + threadIdx.x;
    if (i >= 4 * 8 * 49) return;
    int h   = i / (8 * 49);
    int rem = i - h * (8 * 49);
    int ic  = rem / 49;
    int p   = rem - ic * 49;
    int a = p / 7, bb = p - (p / 7) * 7;
    for (int t = 0; t < h; ++t) { int na = bb, nb = 6 - a; a = na; bb = nb; }  // rot90 mapping
    int pool = ic >> 2, r = ic & 3, rsrc = (r - h) & 3;
    sa_h[i] = sa_w[((pool * R4 + rsrc) * 7 + a) * 7 + bb];
}

// ---------------- 6) spatial attention: 7x7 conv (pad 3) over [mean,max] maps ----------------
__global__ void k_spatial_att(const float* __restrict__ u_avg, const float* __restrict__ u_max,
                              const float* __restrict__ sa_h, float* __restrict__ sp_att) {
    const int i = blockIdx.x * 256 + threadIdx.x;
    const int N = B_ * R4 * HW;
    if (i >= N) return;
    int b   = i / (R4 * HW);
    int rem = i - b * (R4 * HW);
    int h   = rem / HW;
    int hw  = rem - h * HW;
    int yy  = hw / H_, xx = hw - yy * H_;
    const float* wk = sa_h + h * 8 * 49;
    float acc = 0.f;
    for (int ic = 0; ic < 8; ++ic) {
        int pool = ic >> 2, r = ic & 3;
        const float* up = (pool == 0 ? u_avg : u_max) + ((size_t)b * R4 + r) * HW;
        for (int ky = 0; ky < 7; ++ky) {
            int iy = yy + ky - 3;
            if (iy < 0 || iy >= H_) continue;
            for (int kx = 0; kx < 7; ++kx) {
                int ix = xx + kx - 3;
                if (ix < 0 || ix >= H_) continue;
                acc += up[iy * H_ + ix] * wk[ic * 49 + ky * 7 + kx];
            }
        }
    }
    sp_att[i] = sigmoidf_(acc);
}

// ---------------- 7) roll/rot90 conv weights, pack bf16 as Wb[h][o][k] ----------------
__global__ void k_w_transform(const float* __restrict__ conv_w, unsigned short* __restrict__ wb) {
    const int i = blockIdx.x * 256 + threadIdx.x;
    const int N = R4 * O_ * K_TOT;
    if (i >= N) return;
    int h   = i / (O_ * K_TOT);
    int rem = i - h * (O_ * K_TOT);
    int o   = rem / K_TOT;
    int k   = rem - o * K_TOT;
    int cr = k / 9, s = k - cr * 9;
    int c = cr >> 2, r = cr & 3;
    int a = s / 3, bb = s - (s / 3) * 3;
    for (int t = 0; t < h; ++t) { int na = bb, nb = 2 - a; a = na; bb = nb; }  // rot90 on 3x3
    int rsrc = (r - h) & 3;
    wb[i] = f2bf(conv_w[(((size_t)o * C_ + c) * R4 + rsrc) * 9 + a * 3 + bb]);
}

// ---------------- 8) attention-gated implicit-GEMM conv with bf16 WMMA ----------------
// Double-buffered LDS tiles, async B staging, one barrier per K-step.
__global__ void __launch_bounds__(256)
k_att_conv_wmma(const float* __restrict__ y, const float* __restrict__ ch_att,
                const float* __restrict__ sp_att, const unsigned short* __restrict__ wb,
                float* __restrict__ out) {
    constexpr int AS = 40;                          // padded row stride (halves), 80B (16B-aligned)
    constexpr int BS = 40;
    __shared__ __align__(16) float spatt_s[HW];
    __shared__ __align__(16) float chatt_s[C_ * R4];
    __shared__ __align__(16) unsigned short As[2][128 * AS];
    __shared__ __align__(16) unsigned short Bs[2][O_ * BS];

    const int pt  = blockIdx.x;                     // pixel tile 0..31 (128 pixels each)
    const int h   = blockIdx.y;                     // output rotation
    const int b   = blockIdx.z;                     // batch
    const int tid = threadIdx.x;

    const int pb   = pt * 128;
    const int wave = tid >> 5;
    const int lane = tid & 31;
    const int half = lane >> 4;
    const int l16  = lane & 15;

    const float* yb = y + (size_t)b * C_ * R4 * HW;
    const unsigned short* wbh = wb + (size_t)h * O_ * K_TOT;

    // ---- prologue: async-copy attention maps into LDS ----
    {
        const float* sp = sp_att + ((size_t)b * R4 + h) * HW;
        const unsigned l0 = lds_off(spatt_s);
        for (int i = tid; i < HW / 4; i += 256)          // 1089 x 16B
            async_copy16(l0 + i * 16, sp + i * 4);
        const float* ca = ch_att + (size_t)(b * R4 + h) * (C_ * R4);
        const unsigned l1 = lds_off(chatt_s);
        if (tid < (C_ * R4) / 4)                          // 96 x 16B
            async_copy16(l1 + tid * 16, ca + tid * 4);
    }
    wait_async0();
    __syncthreads();

    // ---- staging helpers ----
    auto stageA = [&](int kk, unsigned short* dst) {     // gated bf16 im2col, 2 k per thread
        #pragma unroll
        for (int e = 0; e < 8; ++e) {
            int j   = e * 256 + tid;                     // 0..2047
            int kp  = j >> 7;                            // 0..15 (wave-uniform)
            int pix = j & 127;
            int p   = pb + pix;
            int oy  = p >> 6, ox = p & 63;
            unsigned packed = 0;
            #pragma unroll
            for (int q = 0; q < 2; ++q) {
                int k  = kk + kp * 2 + q;
                int cr = k / 9;
                int s  = k - cr * 9;
                int ky = s / 3, kx = s - ky * 3;
                int iy = oy + ky, ix = ox + kx;
                float v  = yb[(size_t)cr * HW + iy * H_ + ix];
                float gv = v * chatt_s[cr] * spatt_s[iy * H_ + ix];
                packed |= (unsigned)f2bf(gv) << (q * 16);
            }
            *(unsigned*)(dst + pix * AS + kp * 2) = packed;
        }
    };
    auto stageB = [&](int kk, unsigned ldsbase) {        // async 48x32 bf16 weight tile
        if (tid < O_ * 4) {                              // 192 x 16B
            int o = tid >> 2, part = tid & 3;
            async_copy16(ldsbase + o * (BS * 2) + part * 16,
                         wbh + (size_t)o * K_TOT + kk + part * 8);
        }
    };
    const unsigned bs_l[2] = { lds_off(Bs[0]), lds_off(Bs[1]) };

    v8f acc0 = {}; v8f acc1 = {}; v8f acc2 = {};

    stageA(0, As[0]);
    stageB(0, bs_l[0]);
    wait_async0();                                        // B(0) landed (barrier below publishes)

    int cur = 0;
    for (int kk = 0; kk < K_TOT; kk += 32) {              // 108 steps
        __syncthreads();                                  // publish tiles for `cur`
        if (kk + 32 < K_TOT) {                            // stage next into the other buffer
            stageB(kk + 32, bs_l[cur ^ 1]);
            stageA(kk + 32, As[cur ^ 1]);
        }
        union Frag { v16bf v; v4u q[2]; };
        Frag a, b0, b1, b2;
        {   // A: lane<16 -> K 0..15; lane>=16 -> K 16..31 (contiguous)
            const unsigned short* ap = &As[cur][(wave * 16 + l16) * AS + half * 16];
            a.q[0] = *(const v4u*)ap;
            a.q[1] = *(const v4u*)(ap + 8);
        }
        {   // B: lane<16 -> K 0..15; lane>=16 -> K 16..31 (contiguous)
            const unsigned short* bp = &Bs[cur][l16 * BS + half * 16];
            b0.q[0] = *(const v4u*)bp;              b0.q[1] = *(const v4u*)(bp + 8);
            b1.q[0] = *(const v4u*)(bp + 16 * BS);  b1.q[1] = *(const v4u*)(bp + 16 * BS + 8);
            b2.q[0] = *(const v4u*)(bp + 32 * BS);  b2.q[1] = *(const v4u*)(bp + 32 * BS + 8);
        }
        acc0 = __builtin_amdgcn_wmma_f32_16x16x32_bf16(false, a.v, false, b0.v, (short)0, acc0, false, false);
        acc1 = __builtin_amdgcn_wmma_f32_16x16x32_bf16(false, a.v, false, b1.v, (short)0, acc1, false, false);
        acc2 = __builtin_amdgcn_wmma_f32_16x16x32_bf16(false, a.v, false, b2.v, (short)0, acc2, false, false);
        wait_async0();                                    // next B tile landed before next barrier
        cur ^= 1;
    }

    // Epilogue: C/D layout -> VGPR i is pixel row (half selects +8), lane%16 is out-channel.
    const int m0 = wave * 16 + half * 8;
    float* ob = out + ((size_t)(b * 144) * R4 + h) * OPIX;
    #pragma unroll
    for (int t = 0; t < 3; ++t) {
        v8f acc = (t == 0) ? acc0 : ((t == 1) ? acc1 : acc2);
        int o = t * 16 + l16;
        float* op = ob + (size_t)o * R4 * OPIX;
        #pragma unroll
        for (int i = 0; i < 8; ++i)
            op[pb + m0 + i] = acc[i];
    }
}

// ---------------- 9) dense-concat tail: out[:,48:144] = x[:,:,:,1:65,1:65] ----------------
__global__ void k_copy_crop(const float* __restrict__ x, float* __restrict__ out) {
    const int i = blockIdx.x * 256 + threadIdx.x;
    const int N = B_ * C_ * R4 * OPIX;
    if (i >= N) return;
    int pix  = i & (OPIX - 1);
    int rest = i >> 12;
    int r  = rest & 3;
    int bc = rest >> 2;
    int c  = bc % C_;
    int b  = bc / C_;
    int oy = pix >> 6, ox = pix & 63;
    float v = x[(((size_t)(b * C_ + c)) * R4 + r) * HW + (oy + 1) * H_ + (ox + 1)];
    out[(((size_t)(b * 144) + 48 + c) * R4 + r) * OPIX + pix] = v;
}

extern "C" void kernel_launch(void* const* d_in, const int* in_sizes, int n_in,
                              void* d_out, int out_size, void* d_ws, size_t ws_size,
                              hipStream_t stream) {
    (void)in_sizes; (void)n_in; (void)out_size; (void)ws_size;
    const float* x      = (const float*)d_in[0];
    const float* gamma  = (const float*)d_in[1];
    const float* beta   = (const float*)d_in[2];
    const float* fc1    = (const float*)d_in[3];
    const float* fc2    = (const float*)d_in[4];
    const float* sa_w   = (const float*)d_in[5];
    const float* conv_w = (const float*)d_in[6];
    float* out = (float*)d_out;

    float* w = (float*)d_ws;                                   // all region sizes are multiples of 4 floats
    float* mu     = w; w += 96;
    float* rs     = w; w += 96;
    float* y      = w; w += (size_t)B_ * C_ * R4 * HW;         // 26,763,264
    float* s_avg  = w; w += B_ * C_ * R4;                      // 6144
    float* s_max  = w; w += B_ * C_ * R4;
    float* u_avg  = w; w += B_ * R4 * HW;                      // 278,784
    float* u_max  = w; w += B_ * R4 * HW;
    float* ch_att = w; w += B_ * R4 * C_ * R4;                 // 24,576
    float* sa_h   = w; w += 4 * 8 * 49;                        // 1568
    float* sp_att = w; w += B_ * R4 * HW;
    unsigned short* wb = (unsigned short*)w;                   // 4*48*3456 bf16, 16B-aligned

    k_bn_stats    <<<C_,                          256, 0, stream>>>(x, mu, rs);
    k_bn_relu_pool<<<B_ * C_ * R4,                256, 0, stream>>>(x, mu, rs, gamma, beta, y, s_avg, s_max);
    k_channel_pool<<<(B_ * R4 * HW + 255) / 256,  256, 0, stream>>>(y, u_avg, u_max);
    k_channel_att <<<B_ * R4,                     256, 0, stream>>>(s_avg, s_max, fc1, fc2, ch_att);
    k_sa_transform<<<(4 * 8 * 49 + 255) / 256,    256, 0, stream>>>(sa_w, sa_h);
    k_spatial_att <<<(B_ * R4 * HW + 255) / 256,  256, 0, stream>>>(u_avg, u_max, sa_h, sp_att);
    k_w_transform <<<(R4 * O_ * K_TOT + 255) / 256, 256, 0, stream>>>(conv_w, wb);
    dim3 g(OPIX / 128, R4, B_);                                // 32 x 4 x 16
    k_att_conv_wmma<<<g,                          256, 0, stream>>>(y, ch_att, sp_att, wb, out);
    k_copy_crop   <<<(B_ * C_ * R4 * OPIX + 255) / 256, 256, 0, stream>>>(x, out);
}
